// Block_23338852286694
// MI455X (gfx1250) — matrix operand order repaired and checked by reading, not compiled, not run on hardware
//
#include <hip/hip_runtime.h>
#include <hip/hip_bf16.h>

// ---------------------------------------------------------------------------
// Problem constants (from reference)
// ---------------------------------------------------------------------------
#define BB   4
#define SS   1024
#define SEE  1024
#define EE   768
#define HH   12
#define DD   64
#define MM   100
#define NTOK (BB * SS)          // 4096 token rows

typedef __attribute__((ext_vector_type(16))) _Float16 v16h;
typedef __attribute__((ext_vector_type(8)))  _Float16 v8h;
typedef __attribute__((ext_vector_type(8)))  float    v8f;

// ---------------------------------------------------------------------------
// Generic batched GEMM via V_WMMA_F32_16X16X32_F16.
//   C[b,h] = alpha * A[b,h] @ B[b,h](^T) + bias
//
// REQUIREMENTS (hold for every call in this network): M%128==0, N%64==0,
// K%32==0 (and K>=64), row strides multiples of 4 floats -> no bounds checks,
// all global traffic is unconditional b128 loads.
//
// 256 threads = 8 waves. Workgroup tile 128(M) x 64(N). Wave tile 32x32:
// 2x2 grid of 16x16 f32 accumulators, fragments reused across the grid
// (4 WMMAs per wave per 32-K step). f32 -> f16 staged through double-buffered
// LDS: one barrier per K-step, next tile's global loads overlap current WMMAs,
// plus a one-tile-ahead clamped global_prefetch.
// Per-batch pointer offsets: off = (z / Hb)*offXb + (z % Hb)*offXh.
// ---------------------------------------------------------------------------
__global__ __launch_bounds__(256)
void gemm_wmma_kernel(const float* __restrict__ A, const float* __restrict__ Bm,
                      const float* __restrict__ bias, float* __restrict__ C,
                      int M, int N, int K, int lda, int ldb, int ldc,
                      long offAb, long offAh, long offBb, long offBh,
                      long offCb, long offCh, int Hb, float alpha, int transB)
{
    __shared__ _Float16 As[2][128][40];  // [buf][m][k], 16B-aligned rows
    __shared__ _Float16 Bs[2][64][40];   // [buf][n][k]

    const int z  = blockIdx.z;
    const int bz = z / Hb;
    const int hz = z % Hb;
    A  += bz * offAb + hz * offAh;
    Bm += bz * offBb + hz * offBh;
    C  += bz * offCb + hz * offCh;

    const int m0   = blockIdx.y * 128;
    const int n0   = blockIdx.x * 64;
    const int tid  = threadIdx.x;
    const int wave = tid >> 5;        // 0..7
    const int lane = tid & 31;
    const int wm   = wave & 3;        // 32-row M subtile
    const int wn   = wave >> 2;       // 32-col N subtile

    // staging coordinates (loop-invariant)
    const int arow = tid >> 1;              // 0..127
    const int akb  = (tid & 1) * 16;        // 0 or 16
    const int tn   = tid >> 2;              // 0..63   (transB staging)
    const int tkb  = (tid & 3) * 8;
    const int nkk  = tid >> 3;              // 0..31   (non-trans staging)
    const int nnb  = (tid & 7) * 8;

    // stage one 32-wide K tile (k0) into LDS buffer `buf`
    auto stage = [&](int k0, int buf) {
        // prefetch one more tile ahead, clamped in-bounds (branchless)
        const int kp = (k0 + 32 <= K - 32) ? (k0 + 32) : k0;
        // ---- A tile (128 x 32): 16 consecutive floats per thread ----
        {
            const float* pa = A + (long)(m0 + arow) * lda + k0 + akb;
            __builtin_prefetch(A + (long)(m0 + arow) * lda + kp + akb, 0, 1);
            const float4 u0 = *(const float4*)(pa + 0);
            const float4 u1 = *(const float4*)(pa + 4);
            const float4 u2 = *(const float4*)(pa + 8);
            const float4 u3 = *(const float4*)(pa + 12);
            v8h h0, h1;
            h0[0] = (_Float16)u0.x; h0[1] = (_Float16)u0.y;
            h0[2] = (_Float16)u0.z; h0[3] = (_Float16)u0.w;
            h0[4] = (_Float16)u1.x; h0[5] = (_Float16)u1.y;
            h0[6] = (_Float16)u1.z; h0[7] = (_Float16)u1.w;
            h1[0] = (_Float16)u2.x; h1[1] = (_Float16)u2.y;
            h1[2] = (_Float16)u2.z; h1[3] = (_Float16)u2.w;
            h1[4] = (_Float16)u3.x; h1[5] = (_Float16)u3.y;
            h1[6] = (_Float16)u3.z; h1[7] = (_Float16)u3.w;
            *(v8h*)(&As[buf][arow][akb])     = h0;
            *(v8h*)(&As[buf][arow][akb + 8]) = h1;
        }
        // ---- B tile (64 n x 32 k) into [n][k] layout ----
        if (transB) {  // Bm is [N x K]: 8 consecutive k per thread, 1 b128 store
            const float* pb = Bm + (long)(n0 + tn) * ldb + k0 + tkb;
            __builtin_prefetch(Bm + (long)(n0 + tn) * ldb + kp + tkb, 0, 1);
            const float4 u0 = *(const float4*)(pb + 0);
            const float4 u1 = *(const float4*)(pb + 4);
            v8h h;
            h[0] = (_Float16)u0.x; h[1] = (_Float16)u0.y;
            h[2] = (_Float16)u0.z; h[3] = (_Float16)u0.w;
            h[4] = (_Float16)u1.x; h[5] = (_Float16)u1.y;
            h[6] = (_Float16)u1.z; h[7] = (_Float16)u1.w;
            *(v8h*)(&Bs[buf][tn][tkb]) = h;
        } else {       // Bm is [K x N]: coalesced along n, scatter-transpose to LDS
            const float* pb = Bm + (long)(k0 + nkk) * ldb + n0 + nnb;
            __builtin_prefetch(Bm + (long)(kp + nkk) * ldb + n0 + nnb, 0, 1);
            const float4 u0 = *(const float4*)(pb + 0);
            const float4 u1 = *(const float4*)(pb + 4);
            _Float16 hv[8];
            hv[0] = (_Float16)u0.x; hv[1] = (_Float16)u0.y;
            hv[2] = (_Float16)u0.z; hv[3] = (_Float16)u0.w;
            hv[4] = (_Float16)u1.x; hv[5] = (_Float16)u1.y;
            hv[6] = (_Float16)u1.z; hv[7] = (_Float16)u1.w;
#pragma unroll
            for (int i = 0; i < 8; ++i) Bs[buf][nnb + i][nkk] = hv[i];
        }
    };

    v8f acc00 = {}, acc01 = {}, acc10 = {}, acc11 = {};

    // fragment coordinates (loop-invariant)
    const int lrow  = lane & 15;
    const int kbase = (lane >> 4) ? 8 : 0;
    const int ra0   = wm * 32 + lrow;
    const int ra1   = ra0 + 16;
    const int kb2   = (lane >> 4) * 16;
    const int nc0   = wn * 32 + lrow;

    const int T = K >> 5;      // number of 32-wide K tiles (>= 2 in this network)
    stage(0, 0);
    __syncthreads();

    for (int t = 0; t < T; ++t) {
        const int cur = t & 1;
        if (t + 1 < T) stage((t + 1) << 5, cur ^ 1);   // overlaps WMMAs below

        // ---- A fragments (16x32 f16 layout: lanes 0-15 K-base 0, 16-31 K-base 8;
        //      VGPR p<4 -> K = kb+2p, p>=4 -> K = kb+16+2(p-4)) ----
        v16h af0, af1;
#pragma unroll
        for (int p = 0; p < 8; ++p) {
            const int kk = (p < 4) ? (kbase + 2 * p) : (kbase + 16 + 2 * (p - 4));
            af0[2 * p]     = As[cur][ra0][kk];
            af0[2 * p + 1] = As[cur][ra0][kk + 1];
            af1[2 * p]     = As[cur][ra1][kk];
            af1[2 * p + 1] = As[cur][ra1][kk + 1];
        }
        // ---- B fragments (32x16: same K order as A: j<8 -> kbase+j, j>=8 -> kbase+16+(j-8)) ----
        v16h bf0, bf1;
#pragma unroll
        for (int j = 0; j < 16; ++j) {
            const int kj = kbase + j + ((j >= 8) ? 8 : 0);
            bf0[j] = Bs[cur][nc0][kj];
            bf1[j] = Bs[cur][nc0 + 16][kj];
        }

        acc00 = __builtin_amdgcn_wmma_f32_16x16x32_f16(false, af0, false, bf0,
                                                       (short)0, acc00, false, false);
        acc01 = __builtin_amdgcn_wmma_f32_16x16x32_f16(false, af0, false, bf1,
                                                       (short)0, acc01, false, false);
        acc10 = __builtin_amdgcn_wmma_f32_16x16x32_f16(false, af1, false, bf0,
                                                       (short)0, acc10, false, false);
        acc11 = __builtin_amdgcn_wmma_f32_16x16x32_f16(false, af1, false, bf1,
                                                       (short)0, acc11, false, false);
        __syncthreads();   // one barrier per K-step (double-buffered)
    }

    // ---- epilogue: C/D layout lanes 0-15 -> M 0..7, lanes 16-31 -> M 8..15 ----
    const int ncol = lane & 15;
    const int mtop = (lane >> 4) * 8;
    const int gn0  = n0 + wn * 32 + ncol;
    const int gn1  = gn0 + 16;
    const float bv0 = bias ? bias[gn0] : 0.0f;
    const float bv1 = bias ? bias[gn1] : 0.0f;
#pragma unroll
    for (int r = 0; r < 8; ++r) {
        const int row0 = m0 + wm * 32 + mtop + r;
        const int row1 = row0 + 16;
        C[(long)row0 * ldc + gn0] = alpha * acc00[r] + bv0;
        C[(long)row0 * ldc + gn1] = alpha * acc01[r] + bv1;
        C[(long)row1 * ldc + gn0] = alpha * acc10[r] + bv0;
        C[(long)row1 * ldc + gn1] = alpha * acc11[r] + bv1;
    }
}

// ---------------------------------------------------------------------------
// LayerNorm over last dim (Cdim). One workgroup (256 threads) per row.
// Input row = base + (r / rowsPerOuter)*outerStride + (r % rowsPerOuter)*Cdim
// ---------------------------------------------------------------------------
__global__ __launch_bounds__(256)
void layernorm_kernel(const float* __restrict__ x, long outerStride, int rowsPerOuter,
                      const float* __restrict__ g, const float* __restrict__ b,
                      float* __restrict__ y, int Cdim)
{
    __shared__ float red[256];
    const int row = blockIdx.x;
    const float* xr = x + (long)(row / rowsPerOuter) * outerStride
                        + (long)(row % rowsPerOuter) * Cdim;
    float* yr = y + (long)row * Cdim;
    const int tid = threadIdx.x;

    float s = 0.0f;
    for (int j = tid; j < Cdim; j += 256) s += xr[j];
    red[tid] = s; __syncthreads();
    for (int o = 128; o > 0; o >>= 1) { if (tid < o) red[tid] += red[tid + o]; __syncthreads(); }
    const float mu = red[0] / (float)Cdim;
    __syncthreads();

    float v = 0.0f;
    for (int j = tid; j < Cdim; j += 256) { const float d = xr[j] - mu; v += d * d; }
    red[tid] = v; __syncthreads();
    for (int o = 128; o > 0; o >>= 1) { if (tid < o) red[tid] += red[tid + o]; __syncthreads(); }
    const float rstd = rsqrtf(red[0] / (float)Cdim + 1e-5f);
    __syncthreads();

    for (int j = tid; j < Cdim; j += 256)
        yr[j] = g[j] * (xr[j] - mu) * rstd + b[j];
}

// ---------------------------------------------------------------------------
// Row softmax, length L. Optional mask (uint8, [batch][L]); mask==1 replaces
// the score with -10000 before softmax (reference semantics).
// ---------------------------------------------------------------------------
__global__ __launch_bounds__(256)
void softmax_kernel(float* __restrict__ s, int L,
                    const unsigned char* __restrict__ mask, int rowsPerBatch)
{
    __shared__ float red[256];
    const long row = blockIdx.x;
    float* r = s + row * (long)L;
    const unsigned char* mrow =
        mask ? (mask + (long)(row / rowsPerBatch) * L) : (const unsigned char*)nullptr;
    const int tid = threadIdx.x;

    float mx = -3.4e38f;
    for (int j = tid; j < L; j += 256) {
        float v = r[j];
        if (mrow && mrow[j]) v = -10000.0f;
        r[j] = v;
        mx = fmaxf(mx, v);
    }
    red[tid] = mx; __syncthreads();
    for (int o = 128; o > 0; o >>= 1) { if (tid < o) red[tid] = fmaxf(red[tid], red[tid + o]); __syncthreads(); }
    mx = red[0]; __syncthreads();

    float sum = 0.0f;
    for (int j = tid; j < L; j += 256) { const float e = expf(r[j] - mx); r[j] = e; sum += e; }
    red[tid] = sum; __syncthreads();
    for (int o = 128; o > 0; o >>= 1) { if (tid < o) red[tid] += red[tid + o]; __syncthreads(); }
    const float inv = 1.0f / red[0];
    __syncthreads();
    for (int j = tid; j < L; j += 256) r[j] *= inv;
}

// ---------------------------------------------------------------------------
// Memory attention (K or N = 100: tiny, VALU dot kernels)
// ---------------------------------------------------------------------------
__global__ void memkv_kernel(const float* __restrict__ mf, const float* __restrict__ w,
                             const float* __restrict__ bias, float* __restrict__ out)
{
    const int idx = blockIdx.x * blockDim.x + threadIdx.x;
    if (idx >= MM * 2 * EE) return;
    const int m = idx / (2 * EE);
    const int n = idx % (2 * EE);
    float s = bias[n];
    for (int k = 0; k < EE; ++k) s += mf[m * EE + k] * w[(long)k * (2 * EE) + n];
    out[idx] = s;
}

__global__ void memscore_kernel(const float* __restrict__ qkv,
                                const float* __restrict__ memkv, float* __restrict__ wm)
{
    const long idx = (long)blockIdx.x * blockDim.x + threadIdx.x;
    if (idx >= (long)BB * HH * SS * MM) return;
    const int  m  = (int)(idx % MM);
    const long t  = idx / MM;          // (b*H+h)*S + s
    const int  zz = (int)(t >> 10);
    const int  sp = (int)(t & 1023);
    const int  b  = zz / HH, h = zz % HH;
    const float* q  = qkv + ((long)(b * SS + sp)) * (3 * EE) + h * DD;
    const float* mk = memkv + (long)m * (2 * EE) + h * DD;
    float acc = 0.0f;
#pragma unroll
    for (int d = 0; d < DD; ++d) acc += q[d] * mk[d];
    wm[idx] = acc;
}

__global__ void memav_kernel(const float* __restrict__ wm,
                             const float* __restrict__ memkv, float* __restrict__ a1)
{
    const long idx = (long)blockIdx.x * blockDim.x + threadIdx.x;
    if (idx >= (long)NTOK * EE) return;
    const int  e = (int)(idx % EE);
    const long t = idx / EE;           // b*S + s
    const int  h = e / DD, d = e % DD;
    const int  b = (int)(t >> 10);
    const int  sp = (int)(t & 1023);
    const float* wr = wm + ((long)(b * HH + h) * SS + sp) * MM;
    const float* mv = memkv + EE + (long)h * DD + d;
    float acc = 0.0f;
    for (int m = 0; m < MM; ++m) acc += wr[m] * mv[(long)m * (2 * EE)];
    a1[idx] = acc;
}

// ---------------------------------------------------------------------------
// Elementwise helpers
// ---------------------------------------------------------------------------
__global__ void concat_kernel(const float* __restrict__ a, const float* __restrict__ e,
                              float* __restrict__ cat, long rows)
{
    const long idx = (long)blockIdx.x * blockDim.x + threadIdx.x;
    if (idx >= rows * (2 * EE)) return;
    const long r = idx / (2 * EE);
    const int  j = (int)(idx % (2 * EE));
    cat[idx] = (j < EE) ? a[r * EE + j] : e[r * EE + (j - EE)];
}

__global__ void gate_kernel(const float* __restrict__ z, const float* __restrict__ a,
                            const float* __restrict__ e, float* __restrict__ out, long n)
{
    const long i = (long)blockIdx.x * blockDim.x + threadIdx.x;
    if (i >= n) return;
    const float s = 1.0f / (1.0f + expf(-z[i]));
    out[i] = s * a[i] + (1.0f - s) * e[i];
}

__global__ void add_kernel(const float* __restrict__ a, const float* __restrict__ b,
                           float* __restrict__ out, long n)
{
    const long i = (long)blockIdx.x * blockDim.x + threadIdx.x;
    if (i >= n) return;
    out[i] = a[i] + b[i];
}

__global__ void addscale_kernel(const float* __restrict__ a, const float* __restrict__ b,
                                float scale, float* __restrict__ out, long n)
{
    const long i = (long)blockIdx.x * blockDim.x + threadIdx.x;
    if (i >= n) return;
    out[i] = (a[i] + b[i]) * scale;
}

__global__ void gelu_kernel(float* __restrict__ x, long n)
{
    const long i = (long)blockIdx.x * blockDim.x + threadIdx.x;
    if (i >= n) return;
    const float v = x[i];
    x[i] = 0.5f * v * (1.0f + tanhf(0.7978845608028654f * (v + 0.044715f * v * v * v)));
}

// ---------------------------------------------------------------------------
// Host-side orchestration
// ---------------------------------------------------------------------------
static inline long cdiv(long a, long b) { return (a + b - 1) / b; }

extern "C" void kernel_launch(void* const* d_in, const int* in_sizes, int n_in,
                              void* d_out, int out_size, void* d_ws, size_t ws_size,
                              hipStream_t stream)
{
    const float*         x    = (const float*)d_in[0];
    const float*         encf = (const float*)d_in[1];
    const unsigned char* mask = (const unsigned char*)d_in[2];
    const float* ln1_g = (const float*)d_in[3];
    const float* ln1_b = (const float*)d_in[4];
    const float* ln2_g = (const float*)d_in[5];
    const float* ln2_b = (const float*)d_in[6];
    const float* c_attn_w    = (const float*)d_in[7];
    const float* c_attn_b    = (const float*)d_in[8];
    const float* attn_proj_w = (const float*)d_in[9];
    const float* attn_proj_b = (const float*)d_in[10];
    const float* memory_features = (const float*)d_in[11];
    const float* mem_attn_w  = (const float*)d_in[12];
    const float* mem_attn_b  = (const float*)d_in[13];
    const float* mem_alpha_w = (const float*)d_in[14];
    const float* mem_alpha_b = (const float*)d_in[15];
    const float* fcq_w = (const float*)d_in[16];
    const float* fcq_b = (const float*)d_in[17];
    const float* fck_w = (const float*)d_in[18];
    const float* fck_b = (const float*)d_in[19];
    const float* fcv_w = (const float*)d_in[20];
    const float* fcv_b = (const float*)d_in[21];
    const float* enc_proj_w = (const float*)d_in[22];
    const float* enc_proj_b = (const float*)d_in[23];
    const float* fc_alpha1_w = (const float*)d_in[24];
    const float* fc_alpha1_b = (const float*)d_in[25];
    const float* fc_alpha2_w = (const float*)d_in[26];
    const float* fc_alpha2_b = (const float*)d_in[27];
    const float* mlp_fc_w   = (const float*)d_in[28];
    const float* mlp_fc_b   = (const float*)d_in[29];
    const float* mlp_proj_w = (const float*)d_in[30];
    const float* mlp_proj_b = (const float*)d_in[31];

    // ---- workspace carve-up (floats) ----
    float* ws = (float*)d_ws;
    float* LN1   = ws;  ws += (long)NTOK * EE;
    float* QKV   = ws;  ws += (long)NTOK * 3 * EE;
    float* SCR   = ws;  ws += (long)BB * HH * SS * SS;   // 50.3M floats, heavily reused
    float* ASELF = ws;  ws += (long)NTOK * EE;
    float* A1    = ws;  ws += (long)NTOK * EE;
    float* AG    = ws;  ws += (long)NTOK * EE;
    float* ARES  = ws;  ws += (long)NTOK * EE;
    float* QE    = ws;  ws += (long)NTOK * EE;
    float* ENCLN = ws;  ws += (long)NTOK * EE;
    float* KE    = ws;  ws += (long)NTOK * EE;
    float* VE    = ws;  ws += (long)NTOK * EE;
    float* OE    = ws;  ws += (long)NTOK * EE;
    float* E1    = ws;  ws += (long)NTOK * EE;
    float* E2    = ws;  ws += (long)NTOK * EE;
    float* ANEW  = ws;  ws += (long)NTOK * EE;
    float* TMP   = ws;  ws += (long)NTOK * EE;
    // time-disjoint aliases inside SCR:
    float* WM     = SCR;                 // [B*H*S, 100]   (memory-attn scores)
    float* CAT    = SCR;                 // [NTOK, 1536]   (concat for gates)
    float* MLPH   = SCR;                 // [NTOK, 3072]   (MLP hidden)
    float* ZG     = SCR + 20000000L;     // [NTOK, 768]    (gate pre-sigmoid)
    float* MEMKV  = SCR + 45000000L;     // [100, 1536]

    auto gemm = [&](const float* A, const float* Bm, const float* bias, float* C,
                    int M, int N, int K, int lda, int ldb, int ldc,
                    long oAb, long oAh, long oBb, long oBh, long oCb, long oCh,
                    int nbatch, int Hb, float alpha, int transB) {
        dim3 grid((unsigned)cdiv(N, 64), (unsigned)cdiv(M, 128), (unsigned)nbatch);
        gemm_wmma_kernel<<<grid, 256, 0, stream>>>(A, Bm, bias, C, M, N, K, lda, ldb, ldc,
                                                   oAb, oAh, oBb, oBh, oCb, oCh,
                                                   Hb, alpha, transB);
    };
    const long TOKE = (long)NTOK * EE;
    const int  EW   = 256;

    // ================= stage A: self + memory attention =================
    layernorm_kernel<<<NTOK, 256, 0, stream>>>(x, (long)SS * EE, SS, ln1_g, ln1_b, LN1, EE);
    gemm(LN1, c_attn_w, c_attn_b, QKV, NTOK, 3 * EE, EE, EE, 3 * EE, 3 * EE,
         0, 0, 0, 0, 0, 0, 1, 1, 1.0f, 0);
    // scores = q @ k^T (reference has NO 1/sqrt(D) here), batched over (b,h)
    gemm(QKV, QKV + EE, nullptr, SCR, SS, SS, DD, 3 * EE, 3 * EE, SS,
         (long)SS * 3 * EE, DD, (long)SS * 3 * EE, DD,
         (long)HH * SS * SS, (long)SS * SS, BB * HH, HH, 1.0f, 1);
    softmax_kernel<<<BB * HH * SS, 256, 0, stream>>>(SCR, SS, nullptr, 1);
    gemm(SCR, QKV + 2 * EE, nullptr, ASELF, SS, DD, SS, SS, 3 * EE, EE,
         (long)HH * SS * SS, (long)SS * SS, (long)SS * 3 * EE, DD,
         (long)SS * EE, DD, BB * HH, HH, 1.0f, 0);
    // memory attention
    memkv_kernel<<<(int)cdiv(MM * 2 * EE, EW), EW, 0, stream>>>(memory_features, mem_attn_w,
                                                               mem_attn_b, MEMKV);
    memscore_kernel<<<(int)cdiv((long)BB * HH * SS * MM, EW), EW, 0, stream>>>(QKV, MEMKV, WM);
    softmax_kernel<<<BB * HH * SS, 256, 0, stream>>>(WM, MM, nullptr, 1);
    memav_kernel<<<(int)cdiv(TOKE, EW), EW, 0, stream>>>(WM, MEMKV, A1);
    // alpha gate between a and a1
    concat_kernel<<<(int)cdiv((long)NTOK * 2 * EE, EW), EW, 0, stream>>>(ASELF, A1, CAT, NTOK);
    gemm(CAT, mem_alpha_w, mem_alpha_b, ZG, NTOK, EE, 2 * EE, 2 * EE, EE, EE,
         0, 0, 0, 0, 0, 0, 1, 1, 1.0f, 0);
    gate_kernel<<<(int)cdiv(TOKE, EW), EW, 0, stream>>>(ZG, ASELF, A1, AG, TOKE);
    // output proj + residual
    gemm(AG, attn_proj_w, attn_proj_b, TMP, NTOK, EE, EE, EE, EE, EE,
         0, 0, 0, 0, 0, 0, 1, 1, 1.0f, 0);
    add_kernel<<<(int)cdiv(TOKE, EW), EW, 0, stream>>>(x, TMP, ARES, TOKE);

    // ================= stage B: two cross-attentions =================
    layernorm_kernel<<<NTOK, 256, 0, stream>>>(ARES, (long)SS * EE, SS, ln1_g, ln1_b, LN1, EE);
    gemm(LN1, fcq_w, fcq_b, QE, NTOK, EE, EE, EE, EE, EE, 0, 0, 0, 0, 0, 0, 1, 1, 1.0f, 0);

    float* EOUT[2] = { E1, E2 };
    for (int enc = 0; enc < 2; ++enc) {
        layernorm_kernel<<<BB * SEE, 256, 0, stream>>>(encf + (long)enc * SEE * EE,
                                                       (long)2 * SEE * EE, SEE,
                                                       ln1_g, ln1_b, ENCLN, EE);
        gemm(ENCLN, fck_w, fck_b, KE, BB * SEE, EE, EE, EE, EE, EE, 0, 0, 0, 0, 0, 0, 1, 1, 1.0f, 0);
        gemm(ENCLN, fcv_w, fcv_b, VE, BB * SEE, EE, EE, EE, EE, EE, 0, 0, 0, 0, 0, 0, 1, 1, 1.0f, 0);
        // scores = (qe @ ke^T) / sqrt(D)
        gemm(QE, KE, nullptr, SCR, SS, SEE, DD, EE, EE, SEE,
             (long)SS * EE, DD, (long)SEE * EE, DD,
             (long)HH * SS * SEE, (long)SS * SEE, BB * HH, HH, 0.125f, 1);
        softmax_kernel<<<BB * HH * SS, 256, 0, stream>>>(SCR, SEE, mask, HH * SS);
        gemm(SCR, VE, nullptr, OE, SS, DD, SEE, SEE, EE, EE,
             (long)HH * SS * SEE, (long)SS * SEE, (long)SEE * EE, DD,
             (long)SS * EE, DD, BB * HH, HH, 1.0f, 0);
        gemm(OE, enc_proj_w, enc_proj_b, EOUT[enc], NTOK, EE, EE, EE, EE, EE,
             0, 0, 0, 0, 0, 0, 1, 1, 1.0f, 0);
    }
    // gates (in-place into E1/E2)
    concat_kernel<<<(int)cdiv((long)NTOK * 2 * EE, EW), EW, 0, stream>>>(ARES, E1, CAT, NTOK);
    gemm(CAT, fc_alpha1_w, fc_alpha1_b, ZG, NTOK, EE, 2 * EE, 2 * EE, EE, EE,
         0, 0, 0, 0, 0, 0, 1, 1, 1.0f, 0);
    gate_kernel<<<(int)cdiv(TOKE, EW), EW, 0, stream>>>(ZG, ARES, E1, E1, TOKE);
    concat_kernel<<<(int)cdiv((long)NTOK * 2 * EE, EW), EW, 0, stream>>>(ARES, E2, CAT, NTOK);
    gemm(CAT, fc_alpha2_w, fc_alpha2_b, ZG, NTOK, EE, 2 * EE, 2 * EE, EE, EE,
         0, 0, 0, 0, 0, 0, 1, 1, 1.0f, 0);
    gate_kernel<<<(int)cdiv(TOKE, EW), EW, 0, stream>>>(ZG, ARES, E2, E2, TOKE);
    addscale_kernel<<<(int)cdiv(TOKE, EW), EW, 0, stream>>>(E1, E2, 0.70710678118654752f, ANEW, TOKE);

    // ================= stage C: MLP =================
    layernorm_kernel<<<NTOK, 256, 0, stream>>>(ANEW, (long)SS * EE, SS, ln2_g, ln2_b, LN1, EE);
    gemm(LN1, mlp_fc_w, mlp_fc_b, MLPH, NTOK, 4 * EE, EE, EE, 4 * EE, 4 * EE,
         0, 0, 0, 0, 0, 0, 1, 1, 1.0f, 0);
    gelu_kernel<<<(int)cdiv((long)NTOK * 4 * EE, EW), EW, 0, stream>>>(MLPH, (long)NTOK * 4 * EE);
    gemm(MLPH, mlp_proj_w, mlp_proj_b, TMP, NTOK, EE, 4 * EE, 4 * EE, EE, EE,
         0, 0, 0, 0, 0, 0, 1, 1, 1.0f, 0);
    add_kernel<<<(int)cdiv(TOKE, EW), EW, 0, stream>>>(ANEW, TMP, (float*)d_out, TOKE);
}
